// _MahalanobisLoss_71588514889739
// MI455X (gfx1250) — matrix operand order repaired and checked by reading, not compiled, not run on hardware
//
#include <hip/hip_runtime.h>
#include <hip/hip_bf16.h>

// Problem constants (reference: N=131072, D=512)
#define NROWS   131072
#define DDIM    512
#define NC      128               // rows per workgroup chunk
#define NCHUNK  (NROWS / NC)      // 1024
#define LDS_STRIDE 136            // NC + 8 halves: 272B row stride, 68 dwords, %64==4 -> conflict-free b128 tile loads
#define ZT_HALVES (DDIM * LDS_STRIDE)   // 512*136 ushorts = 139264 B

#define PARTIALS_BYTES 4096                       // 1024 floats, d_ws offset 0
#define SYM_BYTES      (DDIM * DDIM * 4)          // 1 MB, d_ws offset 4096

typedef __attribute__((ext_vector_type(16))) __bf16 v16bf;
typedef __attribute__((ext_vector_type(8)))  float  v8f;
typedef __attribute__((ext_vector_type(4)))  unsigned int u32x4;

union V16 {
    v16bf bf;
    u32x4 q[2];
};

// fp32 -> bf16 round-to-nearest-even
__device__ __forceinline__ unsigned int f2bf(float x) {
    unsigned int u = __float_as_uint(x);
    u += 0x7FFFu + ((u >> 16) & 1u);
    return u >> 16;
}

// S = IC + IC^T  (one-time, 1 MB into d_ws; stays L2-resident; symmetric by construction)
__global__ void __launch_bounds__(256)
symmetrize_kernel(const float* __restrict__ inv_cov, float* __restrict__ sym) {
    int idx = blockIdx.x * 256 + threadIdx.x;
    for (int e = idx; e < DDIM * DDIM; e += gridDim.x * 256) {
        int r = e >> 9, c = e & (DDIM - 1);
        sym[e] = inv_cov[e] + inv_cov[c * DDIM + r];
    }
}

// One 16x16 G-tile: acc = sum_k A(i,k)^T * B(j,k) over K = NC rows.
__device__ __forceinline__ v8f gram_tile(const unsigned short* Zt, const V16* a,
                                         int e0, int m, int hi) {
    v8f acc = {};
    #pragma unroll
    for (int kk = 0; kk < NC / 32; ++kk) {
        // B 32x16 layout: lane l<16 -> col N=l, K[0..7]+K[16..23]; lanes 16..31 -> K[8..15]+K[24..31]
        V16 b;
        const int bb = (e0 + m) * LDS_STRIDE + kk * 32 + hi * 8;
        b.q[0] = *(const u32x4*)&Zt[bb];
        b.q[1] = *(const u32x4*)&Zt[bb + 16];
        acc = __builtin_amdgcn_wmma_f32_16x16x32_bf16(
            false, a[kk].bf, false, b.bf, (short)0, acc, false, false);
    }
    return acc;
}

// Tree-shaped <acc_tile, S_tile> contraction for this lane's 8 elements.
__device__ __forceinline__ float tile_dot(v8f acc, float4 s0, float4 s1) {
    float t0 = fmaf(acc[1], s0.y, acc[0] * s0.x);
    float t1 = fmaf(acc[3], s0.w, acc[2] * s0.z);
    float t2 = fmaf(acc[5], s1.y, acc[4] * s1.x);
    float t3 = fmaf(acc[7], s1.w, acc[6] * s1.z);
    return (t0 + t1) + (t2 + t3);
}

template <bool USES>
__global__ void __launch_bounds__(256)
mahalanobis_gram_kernel(const float* __restrict__ feature,
                        const float* __restrict__ mean,
                        const float* __restrict__ inv_cov,
                        const float* __restrict__ sym,   // S = IC+IC^T (USES path)
                        float* __restrict__ partials) {
    extern __shared__ unsigned short Zt[];               // [DDIM][LDS_STRIDE] bf16, transposed chunk
    float* red = (float*)(Zt + ZT_HALVES);               // 256-float reduction scratch

    const int tid  = threadIdx.x;
    const int lane = tid & 31;
    // Wave id is wave-uniform: force into an SGPR so row assignment, loop
    // bounds and row base addresses are scalar (s_cbranch, no exec games,
    // EXEC provably all-ones around WMMA).
    const int wave = __builtin_amdgcn_readfirstlane(tid >> 5);   // 0..7
    const int m    = lane & 15;
    const int hi   = lane >> 4;                          // 0 or 1

    // ---- load-phase thread mapping: 128 col-groups x 2 row-pair slots ----
    const int col4 = (tid & 127) * 4;                    // 0..508
    const int rsel = tid >> 7;                           // 0..1
    const float4 mu = *(const float4*)(mean + col4);     // hoisted: cols fixed per thread

    // Balanced triangular row assignment: wave w owns d-tile rows {w, 31-w, 8+w, 23-w}
    // -> exactly 66 upper-triangular tiles per wave.
    int rows[4];
    rows[0] = wave;
    rows[1] = 31 - wave;
    rows[2] = 8 + wave;
    rows[3] = 23 - wave;

    float partial = 0.0f;

    for (int chunk = blockIdx.x; chunk < NCHUNK; chunk += gridDim.x) {
        const long rowbase = (long)chunk * NC;

        // ================= Phase 1: global -> (center, bf16) -> LDS transposed =========
        #pragma unroll 4
        for (int it = 0; it < NC / 4; ++it) {
            const int r = it * 4 + rsel * 2;             // even row in [0,128)
            const float4 f0 = *(const float4*)(feature + (rowbase + r)     * DDIM + col4);
            const float4 f1 = *(const float4*)(feature + (rowbase + r + 1) * DDIM + col4);
            unsigned int p0 = f2bf(f0.x - mu.x) | (f2bf(f1.x - mu.x) << 16);
            unsigned int p1 = f2bf(f0.y - mu.y) | (f2bf(f1.y - mu.y) << 16);
            unsigned int p2 = f2bf(f0.z - mu.z) | (f2bf(f1.z - mu.z) << 16);
            unsigned int p3 = f2bf(f0.w - mu.w) | (f2bf(f1.w - mu.w) << 16);
            // Zt[d][n]: pair (n=r, r+1) packed in one dword
            *(unsigned int*)&Zt[(col4 + 0) * LDS_STRIDE + r] = p0;
            *(unsigned int*)&Zt[(col4 + 1) * LDS_STRIDE + r] = p1;
            *(unsigned int*)&Zt[(col4 + 2) * LDS_STRIDE + r] = p2;
            *(unsigned int*)&Zt[(col4 + 3) * LDS_STRIDE + r] = p3;
        }
        __syncthreads();

        // ============ Phase 2: upper-triangular G-tiles via WMMA, fused <G,S> ==========
        for (int ri = 0; ri < 4; ++ri) {
            const int i  = rows[ri];                     // scalar
            const int d0 = i * 16;

            // Hoist A fragments for all 4 K-steps (A depends only on (i, k)).
            // 16-bit A 16x32 layout: lane m<16 holds K[0..7]+K[16..23] of row M=m,
            // lanes 16..31 hold K[8..15]+K[24..31].
            V16 a[NC / 32];
            #pragma unroll
            for (int kk = 0; kk < NC / 32; ++kk) {
                const int base = (d0 + m) * LDS_STRIDE + kk * 32 + hi * 8;
                a[kk].q[0] = *(const u32x4*)&Zt[base];
                a[kk].q[1] = *(const u32x4*)&Zt[base + 16];
            }

            if constexpr (USES) {
                // S is symmetric: S[d0+8hi+r][e0+m] == S[e0+m][d0+8hi+r]
                // -> this lane's 8 S values are CONTIGUOUS: two b128 loads per tile.
                int j = i;
                if ((DDIM / 16 - i) & 1) {               // scalar peel for odd trip count
                    const float* sp = sym + (long)(j * 16 + m) * DDIM + d0 + 8 * hi;
                    const float4 s0 = *(const float4*)sp;
                    const float4 s1 = *(const float4*)(sp + 4);
                    v8f acc = gram_tile(Zt, a, j * 16, m, hi);
                    partial = fmaf(tile_dot(acc, s0, s1), 0.5f, partial);  // peeled tile is j==i
                    ++j;
                }
                for (; j < DDIM / 16; j += 2) {
                    // issue both tiles' S loads up-front so they overlap the WMMA chains
                    const float* spa = sym + (long)(j * 16 + m) * DDIM + d0 + 8 * hi;
                    const float* spb = spa + 16L * DDIM;
                    const float4 s0a = *(const float4*)spa;
                    const float4 s1a = *(const float4*)(spa + 4);
                    const float4 s0b = *(const float4*)spb;
                    const float4 s1b = *(const float4*)(spb + 4);
                    // two independent WMMA accumulation chains interleave
                    v8f acc0 = gram_tile(Zt, a, j * 16, m, hi);
                    v8f acc1 = gram_tile(Zt, a, (j + 1) * 16, m, hi);
                    const float wa = (j == i) ? 0.5f : 1.0f;   // scalar select
                    partial = fmaf(tile_dot(acc0, s0a, s1a), wa, partial);
                    partial += tile_dot(acc1, s0b, s1b);
                }
            } else {
                // Fallback (no scratch for S): load IC and IC^T directly.
                for (int j = i; j < DDIM / 16; ++j) {
                    const int e0 = j * 16;
                    v8f acc = gram_tile(Zt, a, e0, m, hi);
                    const float* icp = inv_cov + (long)(d0 + hi * 8) * DDIM + e0 + m;
                    const float* ict = inv_cov + (long)(e0 + m) * DDIM + d0 + hi * 8;
                    float tpart = 0.0f;
                    #pragma unroll
                    for (int r = 0; r < 8; ++r)
                        tpart = fmaf(acc[r], icp[(long)r * DDIM] + ict[r], tpart);
                    partial = fmaf(tpart, (j == i) ? 0.5f : 1.0f, partial);
                }
            }
        }
        __syncthreads();   // protect Zt before next chunk's store phase
    }

    // ---- workgroup reduction ----
    red[tid] = partial;
    __syncthreads();
    for (int s = 128; s > 0; s >>= 1) {
        if (tid < s) red[tid] += red[tid + s];
        __syncthreads();
    }
    if (tid == 0) partials[blockIdx.x] = red[0];
}

__global__ void __launch_bounds__(1024)
final_reduce_kernel(const float* __restrict__ partials, float* __restrict__ out) {
    __shared__ float red[1024];
    const int tid = threadIdx.x;
    red[tid] = partials[tid];
    __syncthreads();
    for (int s = 512; s > 0; s >>= 1) {
        if (tid < s) red[tid] += red[tid + s];
        __syncthreads();
    }
    if (tid == 0) out[0] = red[0] * (1.0f / (float)NROWS);
}

extern "C" void kernel_launch(void* const* d_in, const int* in_sizes, int n_in,
                              void* d_out, int out_size, void* d_ws, size_t ws_size,
                              hipStream_t stream) {
    const float* feature = (const float*)d_in[0];   // [N, D] fp32
    const float* mean    = (const float*)d_in[1];   // [D]    fp32
    const float* inv_cov = (const float*)d_in[2];   // [D, D] fp32
    float* out      = (float*)d_out;
    float* partials = (float*)d_ws;                                   // 1024 floats
    float* sym      = (float*)((char*)d_ws + PARTIALS_BYTES);         // 512x512 fp32

    const bool useS = ws_size >= (size_t)(PARTIALS_BYTES + SYM_BYTES);
    const size_t smem = (size_t)ZT_HALVES * sizeof(unsigned short) + 256 * sizeof(float);

    if (useS) {
        symmetrize_kernel<<<512, 256, 0, stream>>>(inv_cov, sym);
        mahalanobis_gram_kernel<true><<<NCHUNK, 256, smem, stream>>>(
            feature, mean, inv_cov, sym, partials);
    } else {
        mahalanobis_gram_kernel<false><<<NCHUNK, 256, smem, stream>>>(
            feature, mean, inv_cov, nullptr, partials);
    }
    final_reduce_kernel<<<1, 1024, 0, stream>>>(partials, out);
}
